// MultiheadAttention_80290118632171
// MI455X (gfx1250) — matrix operand order repaired and checked by reading, not compiled, not run on hardware
//
#include <hip/hip_runtime.h>
#include <cstdint>
#include <cstddef>

// ---------------------------------------------------------------------------
// MHA forward for MI455X (gfx1250): bf16 WMMA (v_wmma_f32_16x16x32_bf16),
// f32 accumulation, flash-attention second stage.
// Stage pipeline (when ws allows): fp32->bf16 convert kernel, then
// double-buffered GEMM using GLOBAL_LOAD_ASYNC_TO_LDS (ASYNCcnt) when the
// toolchain exposes the gfx1250 async builtins.
// ---------------------------------------------------------------------------

typedef __attribute__((ext_vector_type(16))) __bf16 v16bf;
typedef __attribute__((ext_vector_type(8)))  float  v8f;
typedef int v4i __attribute__((vector_size(16)));   // GCC-style, matches builtin

#define EMBED    1024
#define HEADS    16
#define HEAD_DIM 64
#define BATCH    2
#define SEQ      2048
#define MROWS    (BATCH * SEQ)      // 4096 rows for the QKV GEMMs
#define ATT_SCALE 0.125f            // 1/sqrt(64)

union Frag {
  v16bf v;
  uint4 q[2];
};

__device__ __forceinline__ unsigned short f32_bf16(float f) {
  unsigned int u = __float_as_uint(f);
  unsigned int r = u + 0x7fffu + ((u >> 16) & 1u);   // round-to-nearest-even
  return (unsigned short)(r >> 16);
}

__device__ __forceinline__ v8f v8f_zero() {
  v8f z;
#pragma unroll
  for (int i = 0; i < 8; ++i) z[i] = 0.0f;
  return z;
}

// ---- gfx1250 async global->LDS path (guarded; falls back to sync copies) ---
#if defined(__HIP_DEVICE_COMPILE__) &&                                      \
    __has_builtin(__builtin_amdgcn_global_load_async_to_lds_b128) &&        \
    __has_builtin(__builtin_amdgcn_s_wait_asynccnt)
#define HAVE_ASYNC_LDS 1
#else
#define HAVE_ASYNC_LDS 0
#endif

__device__ __forceinline__ void copy16_g2lds(const unsigned short* g,
                                             unsigned short* l) {
#if HAVE_ASYNC_LDS
  unsigned short* gnc = const_cast<unsigned short*>(g);  // builtin wants non-const
  __builtin_amdgcn_global_load_async_to_lds_b128(
      (__attribute__((address_space(1))) v4i*)gnc,
      (__attribute__((address_space(3))) v4i*)l, 0, 0);
#else
  *(uint4*)l = *(const uint4*)g;
#endif
}

template <int N>
__device__ __forceinline__ void wait_async_le() {
#if HAVE_ASYNC_LDS
  __builtin_amdgcn_s_wait_asynccnt(N);   // N is a compile-time constant
#endif
}

// ---------------------------------------------------------------------------
// Kernel 0: bulk fp32 -> bf16 conversion (vectorized float4 -> 4x bf16).
// ---------------------------------------------------------------------------
__global__ __launch_bounds__(256) void cvt_bf16_kernel(
    const float* __restrict__ src, unsigned short* __restrict__ dst, int n4) {
  const int i = blockIdx.x * blockDim.x + threadIdx.x;
  if (i < n4) {
    const float4 x = ((const float4*)src)[i];
    uint2 p;
    p.x = (unsigned)f32_bf16(x.x) | ((unsigned)f32_bf16(x.y) << 16);
    p.y = (unsigned)f32_bf16(x.z) | ((unsigned)f32_bf16(x.w) << 16);
    ((uint2*)dst)[i] = p;
  }
}

// ---------------------------------------------------------------------------
// Kernel 1 (preferred): C = Xb @ Wb^T + bias with bf16 inputs.
// Double-buffered LDS staging; async global->LDS when available.
// Block: 256 threads (8 waves). Block tile 128x128, wave tile 64x32.
// transposed==0 -> out[b][h][n][d]; ==1 -> out[b][h][d][n]
// ---------------------------------------------------------------------------
__global__ __launch_bounds__(256) void qkv_proj_bf16_kernel(
    const unsigned short* __restrict__ Xb, const unsigned short* __restrict__ Wb,
    const float* __restrict__ bias, unsigned short* __restrict__ out,
    int transposed) {
  const int tid  = threadIdx.x;
  const int lane = tid & 31;
  const int wave = tid >> 5;
  const int hv   = lane >> 4;
  const int l15  = lane & 15;

  const int nBase = blockIdx.x * 128;
  const int mBase = blockIdx.y * 128;
  const int mOff  = (wave >> 2) * 64;
  const int nOff  = (wave & 3) * 32;

  __shared__ unsigned short At[2][128][40];  // double-buffered 128x32 bf16
  __shared__ unsigned short Bt[2][128][40];

  v8f acc[4][2];
#pragma unroll
  for (int mi = 0; mi < 4; ++mi)
#pragma unroll
    for (int ni = 0; ni < 2; ++ni) acc[mi][ni] = v8f_zero();

  // stage one 128x32 bf16 tile pair into buffer bufi (4 x 16B per thread)
  auto stage = [&](int kc, int bufi) {
#pragma unroll
    for (int i = 0; i < 2; ++i) {
      const int f   = tid + i * 256;  // 0..511 chunk index
      const int row = f >> 2;         // 0..127
      const int c8  = (f & 3) * 8;    // 0,8,16,24
      copy16_g2lds(Xb + (size_t)(mBase + row) * EMBED + kc + c8,
                   &At[bufi][row][c8]);
      copy16_g2lds(Wb + (size_t)(nBase + row) * EMBED + kc + c8,
                   &Bt[bufi][row][c8]);
    }
  };

  stage(0, 0);
  for (int c = 0; c < EMBED / 32; ++c) {
    const int bufi = c & 1;
    const bool more = (c + 1) < (EMBED / 32);
    if (more) stage((c + 1) * 32, bufi ^ 1);
    // chunk c's 4 async loads were issued before chunk c+1's; async loads
    // complete in order, so waiting for <=4 outstanding covers chunk c.
    if (more) wait_async_le<4>(); else wait_async_le<0>();
    __syncthreads();

    Frag bf[2];
#pragma unroll
    for (int ni = 0; ni < 2; ++ni) {
      const int n = nOff + ni * 16 + l15;
      bf[ni].q[0] = *(const uint4*)&Bt[bufi][n][hv * 8];
      bf[ni].q[1] = *(const uint4*)&Bt[bufi][n][16 + hv * 8];
    }
#pragma unroll
    for (int mi = 0; mi < 4; ++mi) {
      Frag af;
      const int r = mOff + mi * 16 + l15;
      af.q[0] = *(const uint4*)&At[bufi][r][hv * 8];
      af.q[1] = *(const uint4*)&At[bufi][r][16 + hv * 8];
#pragma unroll
      for (int ni = 0; ni < 2; ++ni)
        acc[mi][ni] = __builtin_amdgcn_wmma_f32_16x16x32_bf16(
            false, af.v, false, bf[ni].v, (short)0, acc[mi][ni], false, false);
    }
    __syncthreads();  // everyone done reading buf before it is refilled
  }

  float bv2[2];
#pragma unroll
  for (int ni = 0; ni < 2; ++ni)
    bv2[ni] = bias[nBase + nOff + ni * 16 + l15];

#pragma unroll
  for (int mi = 0; mi < 4; ++mi) {
#pragma unroll
    for (int ni = 0; ni < 2; ++ni) {
      const int cidx = nBase + nOff + ni * 16 + l15;
      const int h = cidx >> 6;
      const int d = cidx & 63;
#pragma unroll
      for (int r = 0; r < 8; ++r) {
        const float val = acc[mi][ni][r] + bv2[ni];
        const int m = mBase + mOff + mi * 16 + r + 8 * hv;
        const int b = m >> 11;
        const int n = m & 2047;
        size_t idx;
        if (!transposed)
          idx = (((size_t)(b * HEADS + h)) * SEQ + n) * HEAD_DIM + d;
        else
          idx = (((size_t)(b * HEADS + h)) * HEAD_DIM + d) * SEQ + n;
        out[idx] = f32_bf16(val);
      }
    }
  }
}

// ---------------------------------------------------------------------------
// Kernel 1 (fallback, small ws): fused fp32->bf16 conversion inside the GEMM.
// ---------------------------------------------------------------------------
__global__ __launch_bounds__(256) void qkv_proj_kernel(
    const float* __restrict__ X, const float* __restrict__ W,
    const float* __restrict__ bias, unsigned short* __restrict__ out,
    int transposed) {
  const int tid  = threadIdx.x;
  const int lane = tid & 31;
  const int wave = tid >> 5;
  const int hv   = lane >> 4;
  const int l15  = lane & 15;

  const int nBase = blockIdx.x * 128;
  const int mBase = blockIdx.y * 128;
  const int mOff  = (wave >> 2) * 64;
  const int nOff  = (wave & 3) * 32;

  __shared__ unsigned short At[128][40];
  __shared__ unsigned short Bt[128][40];

  v8f acc[4][2];
#pragma unroll
  for (int mi = 0; mi < 4; ++mi)
#pragma unroll
    for (int ni = 0; ni < 2; ++ni) acc[mi][ni] = v8f_zero();

  for (int kc = 0; kc < EMBED; kc += 32) {
#pragma unroll
    for (int i = 0; i < 4; ++i) {
      const int f   = tid + i * 256;
      const int row = f >> 3;
      const int c4  = (f & 7) * 4;
      const float4 xa =
          *(const float4*)(X + (size_t)(mBase + row) * EMBED + kc + c4);
      uint2 pa;
      pa.x = (unsigned)f32_bf16(xa.x) | ((unsigned)f32_bf16(xa.y) << 16);
      pa.y = (unsigned)f32_bf16(xa.z) | ((unsigned)f32_bf16(xa.w) << 16);
      *(uint2*)&At[row][c4] = pa;
      const float4 xb =
          *(const float4*)(W + (size_t)(nBase + row) * EMBED + kc + c4);
      uint2 pb;
      pb.x = (unsigned)f32_bf16(xb.x) | ((unsigned)f32_bf16(xb.y) << 16);
      pb.y = (unsigned)f32_bf16(xb.z) | ((unsigned)f32_bf16(xb.w) << 16);
      *(uint2*)&Bt[row][c4] = pb;
    }
    __syncthreads();

    Frag bf[2];
#pragma unroll
    for (int ni = 0; ni < 2; ++ni) {
      const int n = nOff + ni * 16 + l15;
      bf[ni].q[0] = *(const uint4*)&Bt[n][hv * 8];
      bf[ni].q[1] = *(const uint4*)&Bt[n][16 + hv * 8];
    }
#pragma unroll
    for (int mi = 0; mi < 4; ++mi) {
      Frag af;
      const int r = mOff + mi * 16 + l15;
      af.q[0] = *(const uint4*)&At[r][hv * 8];
      af.q[1] = *(const uint4*)&At[r][16 + hv * 8];
#pragma unroll
      for (int ni = 0; ni < 2; ++ni)
        acc[mi][ni] = __builtin_amdgcn_wmma_f32_16x16x32_bf16(
            false, af.v, false, bf[ni].v, (short)0, acc[mi][ni], false, false);
    }
    __syncthreads();
  }

  float bv2[2];
#pragma unroll
  for (int ni = 0; ni < 2; ++ni)
    bv2[ni] = bias[nBase + nOff + ni * 16 + l15];

#pragma unroll
  for (int mi = 0; mi < 4; ++mi) {
#pragma unroll
    for (int ni = 0; ni < 2; ++ni) {
      const int cidx = nBase + nOff + ni * 16 + l15;
      const int h = cidx >> 6;
      const int d = cidx & 63;
#pragma unroll
      for (int r = 0; r < 8; ++r) {
        const float val = acc[mi][ni][r] + bv2[ni];
        const int m = mBase + mOff + mi * 16 + r + 8 * hv;
        const int b = m >> 11;
        const int n = m & 2047;
        size_t idx;
        if (!transposed)
          idx = (((size_t)(b * HEADS + h)) * SEQ + n) * HEAD_DIM + d;
        else
          idx = (((size_t)(b * HEADS + h)) * HEAD_DIM + d) * SEQ + n;
        out[idx] = f32_bf16(val);
      }
    }
  }
}

// ---------------------------------------------------------------------------
// Kernel 2: flash attention. Q,K bf16 [b][h][n][64]; V bf16 transposed
// [b][h][d][n]. Block: 128 threads (4 waves); each wave owns 16 q rows.
// ---------------------------------------------------------------------------
__global__ __launch_bounds__(128) void attn_kernel(
    const unsigned short* __restrict__ Q, const unsigned short* __restrict__ K,
    const unsigned short* __restrict__ Vt, float* __restrict__ out) {
  const int tid  = threadIdx.x;
  const int lane = tid & 31;
  const int wave = tid >> 5;
  const int hv   = lane >> 4;
  const int l15  = lane & 15;

  const int h = blockIdx.y;
  const int b = blockIdx.z;
  const int qBase = blockIdx.x * 64 + wave * 16;

  const size_t headBase = ((size_t)(b * HEADS + h)) * SEQ * HEAD_DIM;
  const unsigned short* q = Q + headBase;   // [n][64]
  const unsigned short* k = K + headBase;   // [n][64]
  const unsigned short* v = Vt + headBase;  // [d][2048]

  __shared__ unsigned short Pl[4][16][72];

  Frag aQ[2];
  {
    const unsigned short* qr = q + (size_t)(qBase + l15) * HEAD_DIM;
#pragma unroll
    for (int c = 0; c < 2; ++c) {
      aQ[c].q[0] = *(const uint4*)(qr + c * 32 + hv * 8);
      aQ[c].q[1] = *(const uint4*)(qr + c * 32 + 16 + hv * 8);
    }
  }

  float mrow[8], lrow[8];
  v8f O[4];
#pragma unroll
  for (int r = 0; r < 8; ++r) { mrow[r] = -3.0e38f; lrow[r] = 0.0f; }
#pragma unroll
  for (int j = 0; j < 4; ++j) O[j] = v8f_zero();

  for (int t = 0; t < SEQ; t += 64) {
    v8f S[4];
#pragma unroll
    for (int j = 0; j < 4; ++j) S[j] = v8f_zero();
#pragma unroll
    for (int c = 0; c < 2; ++c) {
#pragma unroll
      for (int j = 0; j < 4; ++j) {
        Frag bK;
        const unsigned short* kr =
            k + (size_t)(t + j * 16 + l15) * HEAD_DIM + c * 32 + hv * 8;
        bK.q[0] = *(const uint4*)(kr);
        bK.q[1] = *(const uint4*)(kr + 16);
        S[j] = __builtin_amdgcn_wmma_f32_16x16x32_bf16(
            false, aQ[c].v, false, bK.v, (short)0, S[j], false, false);
      }
    }
#pragma unroll
    for (int j = 0; j < 4; ++j)
#pragma unroll
      for (int r = 0; r < 8; ++r) S[j][r] *= ATT_SCALE;

    float alpha[8];
#pragma unroll
    for (int r = 0; r < 8; ++r) {
      float vm = fmaxf(fmaxf(S[0][r], S[1][r]), fmaxf(S[2][r], S[3][r]));
      vm = fmaxf(vm, __shfl_xor(vm, 1));
      vm = fmaxf(vm, __shfl_xor(vm, 2));
      vm = fmaxf(vm, __shfl_xor(vm, 4));
      vm = fmaxf(vm, __shfl_xor(vm, 8));
      const float mn = fmaxf(mrow[r], vm);
      const float a  = __expf(mrow[r] - mn);
      mrow[r] = mn;
      alpha[r] = a;
      float ps = 0.0f;
#pragma unroll
      for (int j = 0; j < 4; ++j) {
        const float p = __expf(S[j][r] - mn);
        S[j][r] = p;
        ps += p;
      }
      ps += __shfl_xor(ps, 1);
      ps += __shfl_xor(ps, 2);
      ps += __shfl_xor(ps, 4);
      ps += __shfl_xor(ps, 8);
      lrow[r] = lrow[r] * a + ps;
#pragma unroll
      for (int j = 0; j < 4; ++j) O[j][r] *= a;
    }

#pragma unroll
    for (int j = 0; j < 4; ++j)
#pragma unroll
      for (int r = 0; r < 8; ++r)
        Pl[wave][r + 8 * hv][j * 16 + l15] = f32_bf16(S[j][r]);
    __syncthreads();
    Frag aP[2];
#pragma unroll
    for (int c = 0; c < 2; ++c) {
      aP[c].q[0] = *(const uint4*)&Pl[wave][l15][c * 32 + hv * 8];
      aP[c].q[1] = *(const uint4*)&Pl[wave][l15][c * 32 + 16 + hv * 8];
    }
    __syncthreads();

#pragma unroll
    for (int c = 0; c < 2; ++c) {
#pragma unroll
      for (int j = 0; j < 4; ++j) {
        Frag bV;
        const unsigned short* vr =
            v + (size_t)(j * 16 + l15) * SEQ + t + c * 32 + hv * 8;
        bV.q[0] = *(const uint4*)(vr);
        bV.q[1] = *(const uint4*)(vr + 16);
        O[j] = __builtin_amdgcn_wmma_f32_16x16x32_bf16(
            false, aP[c].v, false, bV.v, (short)0, O[j], false, false);
      }
    }
  }

#pragma unroll
  for (int r = 0; r < 8; ++r) {
    const float inv = 1.0f / lrow[r];
    const int qr = qBase + r + 8 * hv;
    float* orow = out + ((size_t)(b * SEQ) + qr) * EMBED + h * HEAD_DIM;
#pragma unroll
    for (int j = 0; j < 4; ++j) orow[j * 16 + l15] = O[j][r] * inv;
  }
}

// ---------------------------------------------------------------------------
// Host launcher
// ---------------------------------------------------------------------------
extern "C" void kernel_launch(void* const* d_in, const int* in_sizes, int n_in,
                              void* d_out, int out_size, void* d_ws,
                              size_t ws_size, hipStream_t stream) {
  (void)in_sizes; (void)n_in; (void)out_size;
  const float* query = (const float*)d_in[0];
  const float* key   = (const float*)d_in[1];
  const float* value = (const float*)d_in[2];
  const float* Wq    = (const float*)d_in[3];
  const float* bq    = (const float*)d_in[4];
  const float* Wk    = (const float*)d_in[5];
  const float* bk    = (const float*)d_in[6];
  const float* Wv    = (const float*)d_in[7];
  const float* bv    = (const float*)d_in[8];
  float* out = (float*)d_out;

  const size_t perQKV = (size_t)BATCH * HEADS * SEQ * HEAD_DIM;  // 4,194,304
  const size_t perW   = (size_t)EMBED * EMBED;                   // 1,048,576

  unsigned short* q_ws = (unsigned short*)d_ws;
  unsigned short* k_ws = q_ws + perQKV;
  unsigned short* v_ws = k_ws + perQKV;

  const size_t need_big = (3 * perQKV + 3 * perQKV + 3 * perW) * sizeof(unsigned short);
  dim3 gA(EMBED / 128, MROWS / 128);  // (8, 32)

  if (ws_size >= need_big) {
    // bf16 staging area behind q/k/v buffers
    unsigned short* xq = v_ws + perQKV;
    unsigned short* xk = xq + perQKV;
    unsigned short* xv = xk + perQKV;
    unsigned short* wqb = xv + perQKV;
    unsigned short* wkb = wqb + perW;
    unsigned short* wvb = wkb + perW;

    const int n4x = (int)(perQKV / 4);  // X tensors: 4096x1024
    const int n4w = (int)(perW / 4);    // W tensors: 1024x1024
    cvt_bf16_kernel<<<(n4x + 255) / 256, 256, 0, stream>>>(query, xq, n4x);
    cvt_bf16_kernel<<<(n4x + 255) / 256, 256, 0, stream>>>(key,   xk, n4x);
    cvt_bf16_kernel<<<(n4x + 255) / 256, 256, 0, stream>>>(value, xv, n4x);
    cvt_bf16_kernel<<<(n4w + 255) / 256, 256, 0, stream>>>(Wq, wqb, n4w);
    cvt_bf16_kernel<<<(n4w + 255) / 256, 256, 0, stream>>>(Wk, wkb, n4w);
    cvt_bf16_kernel<<<(n4w + 255) / 256, 256, 0, stream>>>(Wv, wvb, n4w);

    qkv_proj_bf16_kernel<<<gA, 256, 0, stream>>>(xq, wqb, bq, q_ws, 0);
    qkv_proj_bf16_kernel<<<gA, 256, 0, stream>>>(xk, wkb, bk, k_ws, 0);
    qkv_proj_bf16_kernel<<<gA, 256, 0, stream>>>(xv, wvb, bv, v_ws, 1);
  } else {
    qkv_proj_kernel<<<gA, 256, 0, stream>>>(query, Wq, bq, q_ws, 0);
    qkv_proj_kernel<<<gA, 256, 0, stream>>>(key,   Wk, bk, k_ws, 0);
    qkv_proj_kernel<<<gA, 256, 0, stream>>>(value, Wv, bv, v_ws, 1);
  }

  dim3 gB(SEQ / 64, HEADS, BATCH);    // (32, 16, 2)
  attn_kernel<<<gB, 128, 0, stream>>>(q_ws, k_ws, v_ws, out);
}
